// SuperNet_Layer_55018531062685
// MI455X (gfx1250) — matrix operand order repaired and checked by reading, not compiled, not run on hardware
//
#include <hip/hip_runtime.h>
#include <cstdint>
#include <cstddef>

#ifndef __has_builtin
#define __has_builtin(x) 0
#endif

#if __has_builtin(__builtin_amdgcn_global_load_async_to_lds_b128)
#define USE_ASYNC_LDS 1
#else
#define USE_ASYNC_LDS 0
#endif

typedef __attribute__((ext_vector_type(16))) __bf16 v16bf;
typedef __attribute__((ext_vector_type(8)))  float  v8f;
// GCC-style int 4-vector: matches the async-LDS builtin's parameter type.
typedef int b128_t __attribute__((vector_size(16)));

#define BDIM 8192
#define KDIM 1024
#define NDIM 1024
#define BM   128
#define BN   64
#define BK   32
#define LDA  40        // padded LDS row stride in bf16 elems (80B, 16B-aligned)

// ---------- helpers ----------
__device__ __forceinline__ unsigned short f32_to_bf16(float f) {
    unsigned int u = __float_as_uint(f);
    u += 0x7fffu + ((u >> 16) & 1u);          // round-to-nearest-even
    return (unsigned short)(u >> 16);
}
__device__ __forceinline__ float bf16_to_f32(unsigned short h) {
    return __uint_as_float(((unsigned int)h) << 16);
}
__device__ __forceinline__ float softmax0(const float* sig) {
    return 1.0f / (1.0f + __expf(sig[1] - sig[0]));
}

__device__ __forceinline__ void copy16_async(const unsigned short* g, unsigned short* l) {
#if USE_ASYNC_LDS
    __builtin_amdgcn_global_load_async_to_lds_b128(
        (__attribute__((address_space(1))) b128_t*)g,
        (__attribute__((address_space(3))) b128_t*)l,
        0, 0);
#else
    *(b128_t*)l = *(const b128_t*)g;
#endif
}

__device__ __forceinline__ void wait_async0() {
#if USE_ASYNC_LDS
#if __has_builtin(__builtin_amdgcn_s_wait_asynccnt)
    __builtin_amdgcn_s_wait_asynccnt(0);
#else
    asm volatile("s_wait_asynccnt 0" ::: "memory");
#endif
#endif
}

// ---------- kernel 1: W_core = blend(seed, toeplitz); store W^T as bf16 hi/lo ----------
__global__ void prep_w_split(const float* __restrict__ W_seed,
                             const float* __restrict__ W_sig,
                             unsigned short* __restrict__ Wt_hi,
                             unsigned short* __restrict__ Wt_lo) {
    int idx = blockIdx.x * blockDim.x + threadIdx.x;  // [0, 1M); k contiguous
    int n = idx >> 10;            // OUT index (row of W^T)
    int k = idx & 1023;           // IN index  (col of W^T)
    float w0 = softmax0(W_sig);
    float w1 = 1.0f - w0;
    float seed = W_seed[(size_t)k * NDIM + n];
    // Toeplitz: W_toep[k][n] = (k>=n) ? W_seed[k-n][0] : W_seed[0][n-k]
    float toep = (k >= n) ? W_seed[(size_t)(k - n) << 10] : W_seed[n - k];
    float w = w0 * seed + w1 * toep;
    unsigned short h = f32_to_bf16(w);
    Wt_hi[idx] = h;
    Wt_lo[idx] = f32_to_bf16(w - bf16_to_f32(h));
}

// ---------- kernel 2: split x into bf16 hi/lo (each element split exactly once) ----------
__global__ void prep_x_split(const float* __restrict__ x,
                             unsigned short* __restrict__ x_hi,
                             unsigned short* __restrict__ x_lo) {
    int t = blockIdx.x * blockDim.x + threadIdx.x;    // each handles 4 floats
    float4 v = ((const float4*)x)[t];
    unsigned short h0 = f32_to_bf16(v.x), h1 = f32_to_bf16(v.y);
    unsigned short h2 = f32_to_bf16(v.z), h3 = f32_to_bf16(v.w);
    unsigned short l0 = f32_to_bf16(v.x - bf16_to_f32(h0));
    unsigned short l1 = f32_to_bf16(v.y - bf16_to_f32(h1));
    unsigned short l2 = f32_to_bf16(v.z - bf16_to_f32(h2));
    unsigned short l3 = f32_to_bf16(v.w - bf16_to_f32(h3));
    uint2 hp, lp;
    hp.x = (unsigned)h0 | ((unsigned)h1 << 16);
    hp.y = (unsigned)h2 | ((unsigned)h3 << 16);
    lp.x = (unsigned)l0 | ((unsigned)l1 << 16);
    lp.y = (unsigned)l2 | ((unsigned)l3 << 16);
    ((uint2*)x_hi)[t] = hp;
    ((uint2*)x_lo)[t] = lp;
}

// ---------- kernel 3: tiled GEMM, pure WMMA inner loop ----------
__global__ __launch_bounds__(256) void supernet_gemm(
    const unsigned short* __restrict__ xh,   // [8192,1024] bf16 (K contiguous)
    const unsigned short* __restrict__ xl,
    const unsigned short* __restrict__ wh,   // [1024(N),1024(K)] bf16 = W^T
    const unsigned short* __restrict__ wl,
    const float* __restrict__ b_seed,        // [1024]
    const float* __restrict__ b_sig,         // [2]
    const float* __restrict__ A_sig,         // [2]
    float* __restrict__ out)                 // [8192,1024]
{
    __shared__ __align__(16) unsigned short sAhi[2][BM * LDA];  // 2 x 10 KB
    __shared__ __align__(16) unsigned short sAlo[2][BM * LDA];
    __shared__ __align__(16) unsigned short sBhi[2][BN * LDA];  // 2 x 5 KB
    __shared__ __align__(16) unsigned short sBlo[2][BN * LDA];

    const int tid    = threadIdx.x;
    const int lane   = tid & 31;
    const int waveM  = (tid >> 5) >> 1;   // 0..3
    const int waveN  = (tid >> 5) & 1;    // 0..1
    const int blockM = blockIdx.x * BM;
    const int blockN = blockIdx.y * BN;

    auto stage = [&](int ks, int b) {
        const int k0 = ks * BK;
        #pragma unroll
        for (int t = 0; t < 2; ++t) {                 // A tiles: 128x32 bf16 x2
            int e  = t * 256 + tid;                   // 0..511
            int r  = e >> 2;                          // row 0..127
            int c8 = (e & 3) << 3;                    // k-offset 0/8/16/24
            size_t g = (size_t)(blockM + r) * KDIM + k0 + c8;
            int    o = r * LDA + c8;
            copy16_async(xh + g, &sAhi[b][o]);
            copy16_async(xl + g, &sAlo[b][o]);
        }
        {                                             // B tiles: 64x32 bf16 x2
            int e  = tid;                             // 0..255
            int r  = e >> 2;                          // n-row 0..63
            int c8 = (e & 3) << 3;
            size_t g = (size_t)(blockN + r) * KDIM + k0 + c8;
            int    o = r * LDA + c8;
            copy16_async(wh + g, &sBhi[b][o]);
            copy16_async(wl + g, &sBlo[b][o]);
        }
    };

    v8f acc[2][2] = {};

    stage(0, 0);

    const int NK = KDIM / BK;   // 32
    for (int ks = 0; ks < NK; ++ks) {
        const int b = ks & 1;
        wait_async0();
        __syncthreads();                       // tile visible; prev buffer free
        if (ks + 1 < NK) stage(ks + 1, b ^ 1); // prefetch next K-step

        union Frag { v16bf v; b128_t q[2]; };
        Frag fAh[2], fAl[2], fBh[2], fBl[2];

        // A frag: lane L -> row M=L&15; K runs {kb..kb+7, kb+16..kb+23}, kb=(L>=16)?8:0
        const int kbA  = (lane >> 4) << 3;
        const int mrow = lane & 15;
        #pragma unroll
        for (int mt = 0; mt < 2; ++mt) {
            const unsigned short* ph = &sAhi[b][(waveM * 32 + mt * 16 + mrow) * LDA + kbA];
            const unsigned short* pl = &sAlo[b][(waveM * 32 + mt * 16 + mrow) * LDA + kbA];
            fAh[mt].q[0] = *(const b128_t*)(ph);
            fAh[mt].q[1] = *(const b128_t*)(ph + 16);
            fAl[mt].q[0] = *(const b128_t*)(pl);
            fAl[mt].q[1] = *(const b128_t*)(pl + 16);
        }
        // B frag: lane L -> col N=L&15; K runs {kb2..kb2+7, kb2+16..kb2+23}, kb2=(L>=16)?8:0
        const int kbB = (lane >> 4) << 3;
        #pragma unroll
        for (int nt = 0; nt < 2; ++nt) {
            const int cn = waveN * 32 + nt * 16 + (lane & 15);
            const unsigned short* ph = &sBhi[b][cn * LDA + kbB];
            const unsigned short* pl = &sBlo[b][cn * LDA + kbB];
            fBh[nt].q[0] = *(const b128_t*)(ph);
            fBh[nt].q[1] = *(const b128_t*)(ph + 16);
            fBl[nt].q[0] = *(const b128_t*)(pl);
            fBl[nt].q[1] = *(const b128_t*)(pl + 16);
        }

        // fp32-accurate product via 3-term bf16 split: Ah*Bh + Ah*Bl + Al*Bh
        #pragma unroll
        for (int mt = 0; mt < 2; ++mt)
            #pragma unroll
            for (int nt = 0; nt < 2; ++nt) {
                acc[mt][nt] = __builtin_amdgcn_wmma_f32_16x16x32_bf16(
                    false, fAh[mt].v, false, fBh[nt].v, (short)0, acc[mt][nt], false, false);
                acc[mt][nt] = __builtin_amdgcn_wmma_f32_16x16x32_bf16(
                    false, fAh[mt].v, false, fBl[nt].v, (short)0, acc[mt][nt], false, false);
                acc[mt][nt] = __builtin_amdgcn_wmma_f32_16x16x32_bf16(
                    false, fAl[mt].v, false, fBh[nt].v, (short)0, acc[mt][nt], false, false);
            }
    }

    // ---- fused epilogue: bias mixture + activation mixture ----
    const float bs0 = softmax0(b_sig);         // 'dense' bias weight ('None' = 0)
    const float a0  = softmax0(A_sig);         // 'linear' weight
    const float a1  = 1.0f - a0;               // 'relu' weight
    const int   mOff = (lane >> 4) << 3;       // C layout: lanes>=16 -> M=r+8
    const int   nl   = lane & 15;
    #pragma unroll
    for (int mt = 0; mt < 2; ++mt)
        #pragma unroll
        for (int nt = 0; nt < 2; ++nt) {
            const int ncol = blockN + waveN * 32 + nt * 16 + nl;
            const float bc = bs0 * b_seed[ncol];
            #pragma unroll
            for (int r = 0; r < 8; ++r) {
                const int mrowg = blockM + waveM * 32 + mt * 16 + mOff + r;
                float v = acc[mt][nt][r] + bc;
                out[(size_t)mrowg * NDIM + ncol] = a0 * v + a1 * fmaxf(v, 0.0f);
            }
        }
}

extern "C" void kernel_launch(void* const* d_in, const int* in_sizes, int n_in,
                              void* d_out, int out_size, void* d_ws, size_t ws_size,
                              hipStream_t stream) {
    (void)in_sizes; (void)n_in; (void)out_size; (void)ws_size;
    const float* x      = (const float*)d_in[0];
    const float* W_seed = (const float*)d_in[1];
    const float* b_seed = (const float*)d_in[2];
    const float* W_sig  = (const float*)d_in[3];
    const float* b_sig  = (const float*)d_in[4];
    const float* A_sig  = (const float*)d_in[5];
    float* out = (float*)d_out;

    // Workspace layout (36 MB total):
    //   [0,2MB)    Wt_hi   [2MB,4MB)  Wt_lo
    //   [4MB,20MB) x_hi    [20MB,36MB) x_lo
    char* ws = (char*)d_ws;
    unsigned short* Wt_hi = (unsigned short*)(ws);
    unsigned short* Wt_lo = (unsigned short*)(ws + (size_t)2  * 1024 * 1024);
    unsigned short* x_hi  = (unsigned short*)(ws + (size_t)4  * 1024 * 1024);
    unsigned short* x_lo  = (unsigned short*)(ws + (size_t)20 * 1024 * 1024);

    prep_w_split<<<(KDIM * NDIM) / 256, 256, 0, stream>>>(W_seed, W_sig, Wt_hi, Wt_lo);
    prep_x_split<<<(BDIM * KDIM / 4) / 256, 256, 0, stream>>>(x, x_hi, x_lo);

    dim3 grid(BDIM / BM, NDIM / BN);   // 64 x 16 = 1024 workgroups
    supernet_gemm<<<grid, 256, 0, stream>>>(x_hi, x_lo, Wt_hi, Wt_lo,
                                            b_seed, b_sig, A_sig, out);
}
